// InputFeatureEmbedder_23124103922375
// MI455X (gfx1250) — matrix operand order repaired and checked by reading, not compiled, not run on hardware
//
#include <hip/hip_runtime.h>
#include <hip/hip_bf16.h>

// ---------------------------------------------------------------------------
// AF3 InputFeatureEmbedder for MI455X (gfx1250, wave32, WMMA).
//
// Roofline: ~50 GFLOP total, dominated by the per-pair 4x(128x128) MLP over
// 540x540 pairs.  The naive pair tensor is 149MB f32 touched ~10x (~1.5GB of
// HBM traffic @ 23.3TB/s).  Here the pair tensor is never materialized: each
// workgroup builds 16 pair-vectors in LDS, runs the 4-layer MLP with
// v_wmma_f32_16x16x32_f16 against fragment-packed f16 weights, applies LN and
// the 3 per-block Wbias projections, and emits only the 14MB bias tensor.
// BETA (sequence-local mask) is analytic: allowed(l,m) <=> |m-(15.5+32*(l>>5))|<64.
// All dense GEMMs use f16-in/f32-acc WMMA with weights pre-packed to the
// CDNA5 B-fragment lane layout (contiguous 32B per lane per fragment).
// ---------------------------------------------------------------------------

typedef __attribute__((ext_vector_type(16))) _Float16 v16h;
typedef __attribute__((ext_vector_type(8)))  float    v8f;

#define WMMA_F16(a, b, c) \
  __builtin_amdgcn_wmma_f32_16x16x32_f16(false, (a), false, (b), (short)0, (c), false, false)

enum { EPI_F32 = 0, EPI_BIAS_SIG = 1, EPI_RELU = 2, EPI_GATED = 3 };

// ---- fragment loaders ------------------------------------------------------
// 16-bit A 16x32 (ISA 7.12.2): lane L holds row M=L&15; halves 0..7 -> K =
// base+0..7, halves 8..15 -> K = base+16..23, base = (L>=16 ? 8 : 0).
__device__ inline v16h load_a_frag_g(const _Float16* __restrict__ A, int lda,
                                     int mbase, int kbase, int lane) {
  const _Float16* p = A + (size_t)(mbase + (lane & 15)) * lda + kbase + ((lane >> 4) << 3);
  v16h a;
#pragma unroll
  for (int h = 0; h < 8; ++h) { a[h] = p[h]; a[h + 8] = p[h + 16]; }
  return a;
}

__device__ inline v16h load_a_frag_lds(const _Float16* buf, int kbase, int lane) {
  const _Float16* p = buf + (lane & 15) * 128 + kbase + ((lane >> 4) << 3);
  v16h a;
#pragma unroll
  for (int h = 0; h < 8; ++h) { a[h] = p[h]; a[h + 8] = p[h + 16]; }
  return a;
}

// B fragments are pre-packed: frag f -> 512 contiguous halves, lane-major.
__device__ inline v16h load_b_frag(const _Float16* __restrict__ Bpk, int frag, int lane) {
  return *(const v16h*)(Bpk + (size_t)frag * 512 + (size_t)lane * 16);
}

// ---- weight packing to B-fragment layout -----------------------------------
// 16-bit B 32x16: lane n (0..15) holds column N=n, halves 0..7 -> K=0..7,
// halves 8..15 -> K=16..23; lane n+16 same with K+8.  dst[f*512 + lane*16 + h].
__global__ __launch_bounds__(512)
void pack_b_kernel(const float* __restrict__ W, _Float16* __restrict__ dst,
                   int Korig, int N) {
  const int ntiles = N >> 4;
  const int f  = blockIdx.x;
  const int kb = f / ntiles, nb = f % ntiles;
  const int lane = threadIdx.x >> 4;   // 0..31
  const int h    = threadIdx.x & 15;   // 0..15
  const int k = kb * 32 + (h & 7) + ((h >> 3) << 4) + ((lane >> 4) << 3);
  const int n = nb * 16 + (lane & 15);
  float v = (k < Korig) ? W[(size_t)k * N + n] : 0.f;
  dst[(size_t)f * 512 + (size_t)lane * 16 + h] = (_Float16)v;
}

// ---- generic WMMA GEMM: C[M,N] = A[M,K](f16,row-major) @ Bpk, 4 epilogues --
template <int EPI>
__global__ __launch_bounds__(128)
void gemm_wmma_kernel(const _Float16* __restrict__ A, const _Float16* __restrict__ Bpk,
                      float* __restrict__ C, int M, int N, int K,
                      const float* __restrict__ bias, const float* __restrict__ gate) {
  const int lane = threadIdx.x & 31;
  const int wave = threadIdx.x >> 5;
  const int ntiles = N >> 4;
  const int tile = blockIdx.x * 4 + wave;
  if (tile >= (M >> 4) * ntiles) return;  // wave-uniform exit: EXEC stays all-1s for live waves
  const int nb = tile % ntiles, mb = tile / ntiles;
  v8f acc = {};
  const int ktiles = K >> 5;
  for (int kb = 0; kb < ktiles; ++kb) {
    v16h a = load_a_frag_g(A, K, mb << 4, kb << 5, lane);
    v16h b = load_b_frag(Bpk, kb * ntiles + nb, lane);
    if (kb + 1 < ktiles)
      __builtin_prefetch(Bpk + (size_t)((kb + 1) * ntiles + nb) * 512 + (size_t)lane * 16, 0, 3);
    acc = WMMA_F16(a, b, acc);
  }
  const int col = nb * 16 + (lane & 15);
#pragma unroll
  for (int r = 0; r < 8; ++r) {
    const int row = (mb << 4) + r + ((lane >> 4) << 3);
    const size_t idx = (size_t)row * N + col;
    float v = acc[r];
    if (EPI == EPI_BIAS_SIG) v = 1.f / (1.f + __expf(-(v + bias[col])));
    if (EPI == EPI_RELU)     v = v > 0.f ? v : 0.f;
    if (EPI == EPI_GATED)    v = v * gate[idx];
    C[idx] = v;
  }
}

// ---- fused pair kernel -----------------------------------------------------
// One WG = 16 pairs (i, j0..j0+15): build 128-vec, 4x(128x128) WMMA MLP in
// LDS ping-pong, LN, 12 bias dots (3 blocks x 4 heads) -> biasOut[b][h][i][j].
__global__ __launch_bounds__(256)
void pair_fused_kernel(const float* __restrict__ pos, const int* __restrict__ uid,
                       const int* __restrict__ tok,
                       const float* __restrict__ Wp1, const float* __restrict__ Wp2,
                       const float* __restrict__ Wp3,
                       const float* __restrict__ zsmall,  // [20*20,128]
                       const float* __restrict__ pc1, const float* __restrict__ pc2,
                       const _Float16* __restrict__ Wm_pk,  // 4 layers * 32 frags * 512
                       const float* __restrict__ Wcb,       // [3][128][4]
                       const float* __restrict__ cbv,       // [3][4]
                       float* __restrict__ biasOut) {       // [3][4][540][540]
  __shared__ _Float16 bufA[16 * 128];
  __shared__ _Float16 bufB[16 * 128];
  __shared__ float mu_s[16], rs_s[16];

  const int i  = blockIdx.y;
  const int j0 = blockIdx.x * 16;
  const float pix = pos[i * 3], piy = pos[i * 3 + 1], piz = pos[i * 3 + 2];
  const int ui = uid[i], ti = tok[i];

  for (int e = threadIdx.x; e < 2048; e += 256) {
    const int t = e >> 7, c = e & 127;
    int j = j0 + t; if (j > 539) j = 539;
    const float dx = pix - pos[j * 3], dy = piy - pos[j * 3 + 1], dz = piz - pos[j * 3 + 2];
    const float vm  = (ui == uid[j]) ? 1.f : 0.f;
    const float inv = 1.f / (1.f + dx * dx + dy * dy + dz * dz);
    float val = vm * (dx * Wp1[c] + dy * Wp1[128 + c] + dz * Wp1[256 + c]
                      + inv * Wp2[c] + Wp3[c]);
    val += zsmall[(size_t)(ti * 20 + tok[j]) * 128 + c]
         + pc1[(size_t)i * 128 + c] + pc2[(size_t)j * 128 + c];
    bufA[e] = (_Float16)val;
  }
  __syncthreads();

  const int lane = threadIdx.x & 31;
  const int wave = threadIdx.x >> 5;  // N-tile 0..7
  _Float16* src = bufA;
  _Float16* dst = bufB;
  for (int L = 0; L < 4; ++L) {
    v8f acc = {};
#pragma unroll
    for (int kb = 0; kb < 4; ++kb) {
      v16h a = load_a_frag_lds(src, kb << 5, lane);
      v16h b = load_b_frag(Wm_pk + (size_t)L * 16384, kb * 8 + wave, lane);
      acc = WMMA_F16(a, b, acc);
    }
    const int col = wave * 16 + (lane & 15);
#pragma unroll
    for (int r = 0; r < 8; ++r) {
      const int m = r + ((lane >> 4) << 3);
      float f = acc[r];
      if (L < 3) f = f > 0.f ? f : 0.f;
      dst[m * 128 + col] = (_Float16)f;
    }
    __syncthreads();
    _Float16* tmp = src; src = dst; dst = tmp;
  }

  if (threadIdx.x < 16) {
    float s = 0.f, s2 = 0.f;
    for (int c = 0; c < 128; ++c) { float v = (float)src[threadIdx.x * 128 + c]; s += v; s2 += v * v; }
    const float mu = s * (1.f / 128.f);
    mu_s[threadIdx.x] = mu;
    rs_s[threadIdx.x] = rsqrtf(s2 * (1.f / 128.f) - mu * mu + 1e-5f);
  }
  __syncthreads();

  if (threadIdx.x < 192) {
    const int t = threadIdx.x / 12, r = threadIdx.x % 12;
    const int b = r >> 2, h = r & 3;
    const float mu = mu_s[t], rsg = rs_s[t];
    const float* wc = Wcb + b * 512;
    float acc = cbv[b * 4 + h];
    for (int c = 0; c < 128; ++c)
      acc += ((float)src[t * 128 + c] - mu) * rsg * wc[c * 4 + h];
    const int j = j0 + t;
    if (j < 540)
      biasOut[(((size_t)b * 4 + h) * 540 + i) * 540 + j] = acc;
  }
}

// ---- row LayerNorm (optional affine) -> f16 --------------------------------
__global__ __launch_bounds__(256)
void ln_rows_kernel(const float* __restrict__ x, const float* __restrict__ g,
                    const float* __restrict__ b, _Float16* __restrict__ out,
                    int cols, int hasAffine) {
  __shared__ float s1[256], s2[256];
  const int row = blockIdx.x;
  const float* xr = x + (size_t)row * cols;
  float a0 = 0.f, a1 = 0.f;
  for (int c = threadIdx.x; c < cols; c += 256) { float v = xr[c]; a0 += v; a1 += v * v; }
  s1[threadIdx.x] = a0; s2[threadIdx.x] = a1; __syncthreads();
  for (int s = 128; s > 0; s >>= 1) {
    if (threadIdx.x < s) { s1[threadIdx.x] += s1[threadIdx.x + s]; s2[threadIdx.x] += s2[threadIdx.x + s]; }
    __syncthreads();
  }
  const float mu = s1[0] / (float)cols;
  const float rs = rsqrtf(s2[0] / (float)cols - mu * mu + 1e-5f);
  for (int c = threadIdx.x; c < cols; c += 256) {
    float v = (xr[c] - mu) * rs;
    if (hasAffine) v = v * g[c] + b[c];
    out[(size_t)row * cols + c] = (_Float16)v;
  }
}

// ---- fused adaLN rows: x16 = A1a*LN(noised)+A2a ; t16 = LN_affine(A1t*LN(noised)+A2t)
__global__ __launch_bounds__(128)
void adaln_rows_kernel(const float* __restrict__ noised,
                       const float* __restrict__ A1a, const float* __restrict__ A2a,
                       const float* __restrict__ A1t, const float* __restrict__ A2t,
                       const float* __restrict__ tg, const float* __restrict__ tb,
                       _Float16* __restrict__ x16, _Float16* __restrict__ t16) {
  __shared__ float s1[128], s2[128];
  __shared__ float tpre[384];
  const int row = blockIdx.x;
  const size_t base = (size_t)row * 384;
  float v[3];
  float a0 = 0.f, a1 = 0.f;
#pragma unroll
  for (int u = 0; u < 3; ++u) {
    const int c = threadIdx.x + 128 * u;
    const float x = noised[base + c];
    v[u] = x; a0 += x; a1 += x * x;
  }
  s1[threadIdx.x] = a0; s2[threadIdx.x] = a1; __syncthreads();
  for (int s = 64; s > 0; s >>= 1) {
    if (threadIdx.x < s) { s1[threadIdx.x] += s1[threadIdx.x + s]; s2[threadIdx.x] += s2[threadIdx.x + s]; }
    __syncthreads();
  }
  const float mu  = s1[0] * (1.f / 384.f);
  const float rsg = rsqrtf(s2[0] * (1.f / 384.f) - mu * mu + 1e-5f);
  __syncthreads();
  a0 = 0.f; a1 = 0.f;
#pragma unroll
  for (int u = 0; u < 3; ++u) {
    const int c = threadIdx.x + 128 * u;
    const float xh = (v[u] - mu) * rsg;
    x16[base + c] = (_Float16)(A1a[base + c] * xh + A2a[base + c]);
    const float tp = A1t[base + c] * xh + A2t[base + c];
    tpre[c] = tp; a0 += tp; a1 += tp * tp;
  }
  s1[threadIdx.x] = a0; s2[threadIdx.x] = a1; __syncthreads();
  for (int s = 64; s > 0; s >>= 1) {
    if (threadIdx.x < s) { s1[threadIdx.x] += s1[threadIdx.x + s]; s2[threadIdx.x] += s2[threadIdx.x + s]; }
    __syncthreads();
  }
  const float mu2 = s1[0] * (1.f / 384.f);
  const float rs2 = rsqrtf(s2[0] * (1.f / 384.f) - mu2 * mu2 + 1e-5f);
#pragma unroll
  for (int u = 0; u < 3; ++u) {
    const int c = threadIdx.x + 128 * u;
    t16[base + c] = (_Float16)((tpre[c] - mu2) * rs2 * tg[c] + tb[c]);
  }
}

// ---- single/noised prep ----------------------------------------------------
__global__ __launch_bounds__(256)
void single_prep_kernel(const float* __restrict__ single0, const float* __restrict__ sproj,
                        const int* __restrict__ tok, const float* __restrict__ rl,
                        const float* __restrict__ Wqn,  // [3,384]
                        float* __restrict__ single_f32, _Float16* __restrict__ single16,
                        _Float16* __restrict__ rs16, float* __restrict__ noised_f32) {
  const int idx = blockIdx.x * 256 + threadIdx.x;
  if (idx >= 540 * 384) return;
  const int i = idx / 384, c = idx % 384;
  const float s0 = single0[idx];
  const float s  = s0 + sproj[(size_t)tok[i] * 384 + c];
  const float nz = s0 + rl[i * 3 + 0] * Wqn[c] + rl[i * 3 + 1] * Wqn[384 + c]
                      + rl[i * 3 + 2] * Wqn[768 + c];
  single_f32[idx] = s;
  single16[idx]   = (_Float16)s;
  rs16[idx]       = (_Float16)(s > 0.f ? s : 0.f);
  noised_f32[idx] = nz;
}

// ---- feature concat -> padded f16 [544,416] --------------------------------
__global__ __launch_bounds__(256)
void feat_build_kernel(const float* __restrict__ pos, const float* __restrict__ charge,
                       const float* __restrict__ mask, const float* __restrict__ elem,
                       const float* __restrict__ namec, _Float16* __restrict__ feat16) {
  const int idx = blockIdx.x * 256 + threadIdx.x;
  if (idx >= 544 * 416) return;
  const int i = idx / 416, c = idx % 416;
  float v = 0.f;
  if (i < 540) {
    if (c < 3)        v = pos[i * 3 + c];
    else if (c == 3)  v = charge[i];
    else if (c == 4)  v = mask[i];
    else if (c < 133) v = elem[(size_t)i * 128 + (c - 5)];
    else if (c < 389) v = namec[(size_t)i * 256 + (c - 133)];
  }
  feat16[idx] = (_Float16)v;
}

// ---- attention: one (query i, head h) per WG -------------------------------
__global__ __launch_bounds__(256)
void attn_kernel(const float* __restrict__ Q, const float* __restrict__ Km,
                 const float* __restrict__ V, const float* __restrict__ bias,  // [4][540][540]
                 _Float16* __restrict__ o16) {
  __shared__ float qsh[96];
  __shared__ float lg[540];
  __shared__ float red[256];
  const int i = blockIdx.x, h = blockIdx.y;
  if (threadIdx.x < 96) qsh[threadIdx.x] = Q[(size_t)i * 384 + h * 96 + threadIdx.x];
  __syncthreads();
  const float cen = 15.5f + 32.0f * (float)(i >> 5);
  float lmax = -3.0e38f;
  for (int m = threadIdx.x; m < 540; m += 256) {
    const float* kr = Km + (size_t)m * 384 + h * 96;
    float acc = 0.f;
    for (int d = 0; d < 96; ++d) acc += qsh[d] * kr[d];
    acc = acc * 0.1020620726f + bias[((size_t)h * 540 + i) * 540 + m];
    if (fabsf((float)m - cen) >= 64.f) acc -= 1.0e10f;   // analytic BETA
    lg[m] = acc;
    lmax = fmaxf(lmax, acc);
  }
  red[threadIdx.x] = lmax; __syncthreads();
  for (int s = 128; s > 0; s >>= 1) {
    if (threadIdx.x < s) red[threadIdx.x] = fmaxf(red[threadIdx.x], red[threadIdx.x + s]);
    __syncthreads();
  }
  const float gmax = red[0];
  __syncthreads();
  float ls = 0.f;
  for (int m = threadIdx.x; m < 540; m += 256) { float p = __expf(lg[m] - gmax); lg[m] = p; ls += p; }
  red[threadIdx.x] = ls; __syncthreads();
  for (int s = 128; s > 0; s >>= 1) {
    if (threadIdx.x < s) red[threadIdx.x] += red[threadIdx.x + s];
    __syncthreads();
  }
  const float inv = 1.f / red[0];
  if (threadIdx.x < 96) {
    float acc = 0.f;
    for (int m = 0; m < 540; ++m) acc += lg[m] * V[(size_t)m * 384 + h * 96 + threadIdx.x];
    o16[(size_t)i * 384 + h * 96 + threadIdx.x] = (_Float16)(acc * inv);
  }
}

// ---- misc elementwise ------------------------------------------------------
__global__ __launch_bounds__(256)
void mul_f16_kernel(const float* __restrict__ a, const float* __restrict__ b,
                    _Float16* __restrict__ o, int n) {
  const int idx = blockIdx.x * 256 + threadIdx.x;
  if (idx < n) o[idx] = (_Float16)(a[idx] * b[idx]);
}

__global__ __launch_bounds__(256)
void update_kernel(float* __restrict__ noised, const float* __restrict__ aacc,
                   const float* __restrict__ facc, _Float16* __restrict__ n16, int writeHalf) {
  const int idx = blockIdx.x * 256 + threadIdx.x;
  if (idx >= 540 * 384) return;
  const float nv = noised[idx] + aacc[idx] + facc[idx];
  noised[idx] = nv;
  if (writeHalf) n16[idx] = (_Float16)nv;
}

__global__ __launch_bounds__(512)
void biasprep_kernel(const float* __restrict__ g, const float* __restrict__ bb,
                     const float* __restrict__ wb, float* __restrict__ Wcb,
                     float* __restrict__ cb) {
  const int tid = threadIdx.x;
  if (tid < 512) { const int c = tid >> 2; Wcb[tid] = g[c] * wb[tid]; }
  if (tid < 4) {
    float s = 0.f;
    for (int c = 0; c < 128; ++c) s += bb[c] * wb[c * 4 + tid];
    cb[tid] = s;
  }
}

__global__ __launch_bounds__(256)
void pool_kernel(const float* __restrict__ feats, const float* __restrict__ mask,
                 const float* __restrict__ restype, const float* __restrict__ profile,
                 const float* __restrict__ delmean, float* __restrict__ out) {
  const int t = blockIdx.x;
  float cnt = 0.f;
  for (int a = 0; a < 27; ++a) cnt += (mask[t * 27 + a] > 0.f) ? 1.f : 0.f;
  const float invc = 1.f / fmaxf(cnt, 1.f);
  for (int c = threadIdx.x; c < 768; c += 256) {
    float s = 0.f;
    for (int a = 0; a < 27; ++a)
      if (mask[t * 27 + a] > 0.f) s += feats[(size_t)(t * 27 + a) * 768 + c];
    out[(size_t)t * 833 + c] = s * invc;
  }
  for (int e = threadIdx.x; e < 65; e += 256) {
    float v = (e < 32) ? restype[(size_t)t * 32 + e]
            : (e < 64) ? profile[(size_t)t * 32 + (e - 32)]
                       : delmean[t];
    out[(size_t)t * 833 + 768 + e] = v;
  }
}

// ---------------------------------------------------------------------------
extern "C" void kernel_launch(void* const* d_in, const int* in_sizes, int n_in,
                              void* d_out, int out_size, void* d_ws, size_t ws_size,
                              hipStream_t stream) {
  (void)in_sizes; (void)n_in; (void)out_size; (void)ws_size;
  auto F = [&](int i) { return (const float*)d_in[i]; };
  auto Iv = [&](int i) { return (const int*)d_in[i]; };

  // setup_inputs() dict order, params flattened recursively in insertion order.
  const float* pos     = F(0);
  const float* charge  = F(1);
  const float* mask    = F(2);
  const float* elem    = F(3);
  const float* namec   = F(4);
  const int*   uid     = Iv(5);
  const int*   tokidx  = Iv(6);
  const float* restype = F(7);
  const float* profile = F(8);
  const float* delmean = F(9);
  const float* rl      = F(10);
  const float* strunk  = F(11);
  const float* zij     = F(12);
  const int PB = 13;  // params: W_c,Wp1,Wp2,Wp3,ln_s_g,ln_s_b,W_c_trunk,ln_z_g,
                      //         ln_z_b,W_p_trunk,W_q_noisy,Wp_c1,Wp_c2,W_m1..4,
                      //         blocks(3x22 leaves), W_pool
  auto BL = [&](int b, int l) { return PB + 17 + b * 22 + l; };
  const int iW_pool = PB + 17 + 66;

  // ---- workspace carve-out -------------------------------------------------
  char* ws = (char*)d_ws;
  size_t off = 0;
  auto A8 = [&](size_t bytes) { size_t r = off; off += (bytes + 255) & ~(size_t)255; return r; };
  auto WF = [&](size_t elems) { return (float*)(ws + A8(elems * 4)); };
  auto WH = [&](size_t elems) { return (_Float16*)(ws + A8(elems * 2)); };
  const int NP = 544;  // 540 padded to 16

  _Float16* feat16   = WH((size_t)NP * 416);
  _Float16* sln16    = WH((size_t)32 * 768);
  _Float16* zln16    = WH((size_t)400 * 768);
  _Float16* sn16     = WH((size_t)NP * 384);
  _Float16* single16 = WH((size_t)NP * 384);
  _Float16* rs16     = WH((size_t)NP * 384);
  _Float16* x16      = WH((size_t)NP * 384);
  _Float16* t16      = WH((size_t)NP * 384);
  _Float16* o16      = WH((size_t)NP * 384);
  _Float16* uprod16  = WH((size_t)NP * 1536);
  _Float16* noised16 = WH((size_t)NP * 384);

  float* single0    = WF((size_t)NP * 384);
  float* sproj      = WF((size_t)32 * 384);
  float* zsmall     = WF((size_t)400 * 128);
  float* pc1        = WF((size_t)NP * 128);
  float* pc2        = WF((size_t)NP * 128);
  float* single_f32 = WF((size_t)NP * 384);
  float* noised_f32 = WF((size_t)NP * 384);
  float* A1a = WF((size_t)NP * 384); float* A2a = WF((size_t)NP * 384);
  float* A1t = WF((size_t)NP * 384); float* A2t = WF((size_t)NP * 384);
  float* ga  = WF((size_t)NP * 384); float* gt  = WF((size_t)NP * 384);
  float* qbuf = WF((size_t)NP * 384);
  float* kbuf = WF((size_t)NP * 384);
  float* vbuf = WF((size_t)NP * 384);
  float* u1 = WF((size_t)NP * 1536);
  float* u2 = WF((size_t)NP * 1536);
  float* attn_acc = WF((size_t)NP * 384);
  float* ff_acc   = WF((size_t)NP * 384);
  float* feats    = WF((size_t)NP * 768);
  float* biasbuf  = WF((size_t)3 * 4 * 540 * 540);
  float* Wcb      = WF((size_t)3 * 128 * 4);
  float* cbv      = WF((size_t)3 * 4);

  _Float16* Wc_pk   = WH((size_t)416 * 384);
  _Float16* Wct_pk  = WH((size_t)768 * 384);
  _Float16* Wpt_pk  = WH((size_t)768 * 128);
  _Float16* Wpc1_pk = WH((size_t)384 * 128);
  _Float16* Wpc2_pk = WH((size_t)384 * 128);
  _Float16* Wm_pk   = WH((size_t)4 * 128 * 128);
  _Float16* Wpool_pk = WH((size_t)384 * 768);
  _Float16* bq[3][13];
  for (int b = 0; b < 3; ++b) {
    for (int w = 0; w < 10; ++w) bq[b][w] = WH((size_t)384 * 384);
    bq[b][10] = WH((size_t)384 * 1536);
    bq[b][11] = WH((size_t)384 * 1536);
    bq[b][12] = WH((size_t)1536 * 384);
  }

  // ---- helpers -------------------------------------------------------------
  auto pack = [&](const float* src, _Float16* dst, int Korig, int N) {
    const int K32 = (Korig + 31) & ~31;
    pack_b_kernel<<<(K32 / 32) * (N / 16), 512, 0, stream>>>(src, dst, Korig, N);
  };
  auto gemm = [&](const _Float16* A, const _Float16* B, float* C, int M, int N, int K,
                  int epi, const float* bias, const float* gate) {
    const int blocks = ((M / 16) * (N / 16) + 3) / 4;
    switch (epi) {
      case EPI_F32:      gemm_wmma_kernel<EPI_F32><<<blocks, 128, 0, stream>>>(A, B, C, M, N, K, bias, gate); break;
      case EPI_BIAS_SIG: gemm_wmma_kernel<EPI_BIAS_SIG><<<blocks, 128, 0, stream>>>(A, B, C, M, N, K, bias, gate); break;
      case EPI_RELU:     gemm_wmma_kernel<EPI_RELU><<<blocks, 128, 0, stream>>>(A, B, C, M, N, K, bias, gate); break;
      default:           gemm_wmma_kernel<EPI_GATED><<<blocks, 128, 0, stream>>>(A, B, C, M, N, K, bias, gate); break;
    }
  };

  // ---- pack all WMMA weights ----------------------------------------------
  pack(F(PB + 0), Wc_pk, 389, 384);
  pack(F(PB + 6), Wct_pk, 768, 384);
  pack(F(PB + 9), Wpt_pk, 768, 128);
  pack(F(PB + 11), Wpc1_pk, 384, 128);
  pack(F(PB + 12), Wpc2_pk, 384, 128);
  for (int m = 0; m < 4; ++m) pack(F(PB + 13 + m), Wm_pk + (size_t)m * 128 * 128, 128, 128);
  pack(F(iW_pool), Wpool_pk, 384, 768);
  for (int b = 0; b < 3; ++b) {
    pack(F(BL(b, 0)),  bq[b][0], 384, 384);   // aln_a.W1
    pack(F(BL(b, 2)),  bq[b][1], 384, 384);   // aln_a.W2
    pack(F(BL(b, 3)),  bq[b][2], 384, 384);   // Wg_a
    pack(F(BL(b, 5)),  bq[b][3], 384, 384);   // Wq
    pack(F(BL(b, 6)),  bq[b][4], 384, 384);   // Wk
    pack(F(BL(b, 7)),  bq[b][5], 384, 384);   // Wv
    pack(F(BL(b, 8)),  bq[b][6], 384, 384);   // Wo
    pack(F(BL(b, 12)), bq[b][7], 384, 384);   // aln_t.W1
    pack(F(BL(b, 14)), bq[b][8], 384, 384);   // aln_t.W2
    pack(F(BL(b, 15)), bq[b][9], 384, 384);   // Wg_t
    pack(F(BL(b, 19)), bq[b][10], 384, 1536); // Wup_a
    pack(F(BL(b, 20)), bq[b][11], 384, 1536); // Wup_b
    pack(F(BL(b, 21)), bq[b][12], 1536, 384); // Wdown
    biasprep_kernel<<<1, 512, 0, stream>>>(F(BL(b, 9)), F(BL(b, 10)), F(BL(b, 11)),
                                           Wcb + b * 512, cbv + b * 4);
  }

  // ---- prep ----------------------------------------------------------------
  feat_build_kernel<<<(544 * 416 + 255) / 256, 256, 0, stream>>>(pos, charge, mask, elem, namec, feat16);
  gemm(feat16, Wc_pk, single0, NP, 384, 416, EPI_F32, nullptr, nullptr);
  ln_rows_kernel<<<20, 256, 0, stream>>>(strunk, F(PB + 4), F(PB + 5), sln16, 768, 1);
  gemm(sln16, Wct_pk, sproj, 32, 384, 768, EPI_F32, nullptr, nullptr);
  ln_rows_kernel<<<400, 256, 0, stream>>>(zij, F(PB + 7), F(PB + 8), zln16, 768, 1);
  gemm(zln16, Wpt_pk, zsmall, 400, 128, 768, EPI_F32, nullptr, nullptr);
  single_prep_kernel<<<(540 * 384 + 255) / 256, 256, 0, stream>>>(
      single0, sproj, tokidx, rl, F(PB + 10), single_f32, single16, rs16, noised_f32);
  ln_rows_kernel<<<540, 256, 0, stream>>>(single_f32, nullptr, nullptr, sn16, 384, 0);
  gemm(rs16, Wpc1_pk, pc1, NP, 128, 384, EPI_F32, nullptr, nullptr);
  gemm(rs16, Wpc2_pk, pc2, NP, 128, 384, EPI_F32, nullptr, nullptr);

  // ---- fused pair MLP + per-block bias (never materializes 149MB pair) ----
  pair_fused_kernel<<<dim3(34, 540), 256, 0, stream>>>(
      pos, uid, tokidx, F(PB + 1), F(PB + 2), F(PB + 3), zsmall, pc1, pc2,
      Wm_pk, Wcb, cbv, biasbuf);

  // ---- 3 attention blocks --------------------------------------------------
  for (int b = 0; b < 3; ++b) {
    gemm(sn16, bq[b][0], A1a, NP, 384, 384, EPI_BIAS_SIG, F(BL(b, 1)), nullptr);
    gemm(sn16, bq[b][1], A2a, NP, 384, 384, EPI_F32, nullptr, nullptr);
    gemm(sn16, bq[b][7], A1t, NP, 384, 384, EPI_BIAS_SIG, F(BL(b, 13)), nullptr);
    gemm(sn16, bq[b][8], A2t, NP, 384, 384, EPI_F32, nullptr, nullptr);
    gemm(single16, bq[b][2], ga, NP, 384, 384, EPI_BIAS_SIG, F(BL(b, 4)), nullptr);
    gemm(single16, bq[b][9], gt, NP, 384, 384, EPI_BIAS_SIG, F(BL(b, 16)), nullptr);
    adaln_rows_kernel<<<540, 128, 0, stream>>>(noised_f32, A1a, A2a, A1t, A2t,
                                               F(BL(b, 17)), F(BL(b, 18)), x16, t16);
    gemm(x16, bq[b][3], qbuf, NP, 384, 384, EPI_F32, nullptr, nullptr);
    gemm(x16, bq[b][4], kbuf, NP, 384, 384, EPI_F32, nullptr, nullptr);
    gemm(x16, bq[b][5], vbuf, NP, 384, 384, EPI_F32, nullptr, nullptr);
    attn_kernel<<<dim3(540, 4), 256, 0, stream>>>(qbuf, kbuf, vbuf,
                                                  biasbuf + (size_t)b * 4 * 540 * 540, o16);
    gemm(o16, bq[b][6], attn_acc, NP, 384, 384, EPI_GATED, nullptr, ga);
    gemm(t16, bq[b][10], u1, NP, 1536, 384, EPI_F32, nullptr, nullptr);
    gemm(t16, bq[b][11], u2, NP, 1536, 384, EPI_F32, nullptr, nullptr);
    mul_f16_kernel<<<(NP * 1536 + 255) / 256, 256, 0, stream>>>(u1, u2, uprod16, NP * 1536);
    gemm(uprod16, bq[b][12], ff_acc, NP, 384, 1536, EPI_GATED, nullptr, gt);
    update_kernel<<<(540 * 384 + 255) / 256, 256, 0, stream>>>(noised_f32, attn_acc, ff_acc,
                                                               noised16, (b == 2) ? 1 : 0);
  }

  // ---- pooling + concat ----------------------------------------------------
  gemm(noised16, Wpool_pk, feats, NP, 768, 384, EPI_RELU, nullptr, nullptr);
  pool_kernel<<<20, 256, 0, stream>>>(feats, mask, restype, profile, delmean, (float*)d_out);
}
